// My_Model_13408887898682
// MI455X (gfx1250) — matrix operand order repaired and checked by reading, not compiled, not run on hardware
//
#include <hip/hip_runtime.h>
#include <cstdint>
#include <cstddef>

// ---------------- model dims ----------------
#define B_    4
#define SL_   512
#define WIN_  256
#define SCH_  3
#define CCH_  128
#define F_    8
#define RNN_  128
#define FC_   256
#define L1_   62            // pooled conv1 length
#define L2_   13            // pooled conv2 length
#define D1_   (L2_*CCH_)    // 1664
#define NSEQ  (B_*SL_)      // 2048
#define G3_   (3*RNN_)      // 384

typedef __attribute__((ext_vector_type(16))) _Float16 v16h;
typedef __attribute__((ext_vector_type(8)))  float    v8f;

// ---------------- WMMA fragment helpers (CDNA5 16x16x32 f16 layouts) ----------------
__device__ __forceinline__ int lane_id() { return threadIdx.x & 31; }

// A: row-major (16 x K), lda in halfs.  Lane l holds row l&15.
// half j -> K = kk + (l<16?0:16) + j
__device__ __forceinline__ v16h load_a_frag(const _Float16* A, int lda, int kk) {
  int l = lane_id();
  const _Float16* p = A + (size_t)(l & 15) * lda + kk;
  int hi8 = (l >> 4) * 16;
  union { v16h v; uint4 q[2]; } u;
  u.q[0] = *(const uint4*)(p + hi8);
  u.q[1] = *(const uint4*)(p + 8 + hi8);
  return u.v;
}

// B stored transposed (N x K) row-major ("Bt"), ldb in halfs. Lane l holds col l&15.
// half j -> K = kk + (l<16?0:16) + j  => contiguous 32 bytes.
__device__ __forceinline__ v16h load_bt_frag(const _Float16* Bt, int ldb, int kk) {
  int l = lane_id();
  const _Float16* p = Bt + (size_t)(l & 15) * ldb + kk + (size_t)((l >> 4) * 16);
  union { v16h v; uint4 q[2]; } u;
  u.q[0] = *(const uint4*)(p);
  u.q[1] = *(const uint4*)(p + 8);
  return u.v;
}

// B row-major (K x N), Bc pre-offset to tile's first column, ldb = N stride.
__device__ __forceinline__ v16h load_b_rm_frag(const _Float16* Bc, int ldb, int kk) {
  int l = lane_id();
  const _Float16* p = Bc + (size_t)(kk + (l >> 4) * 16) * ldb + (l & 15);
  v16h v;
  #pragma unroll
  for (int j = 0; j < 16; ++j) v[j] = p[(size_t)j * ldb];
  return v;
}

__device__ __forceinline__ v8f wmma16(v16h a, v16h b, v8f c) {
  return __builtin_amdgcn_wmma_f32_16x16x32_f16(false, a, false, b, (short)0, c, false, false);
}

__device__ __forceinline__ float leaky(float v) { return v > 0.f ? v : 0.2f * v; }

// ---------------- generic WMMA GEMM: C = act(A(f16,MxK) @ B(f16) + bias) ----------------
// BT=true : B stored (N,K) row-major. BT=false: B stored (K,N) row-major.
// block = 128 threads (4 waves), wave w -> 16x16 tile at (blockIdx.x*16, blockIdx.y*64 + w*16)
template<bool BT, int ACT>
__global__ void __launch_bounds__(128)
wmma_gemm(const _Float16* __restrict__ A, const _Float16* __restrict__ B,
          const float* __restrict__ bias, float* __restrict__ C,
          int N, int K, int lda, int ldb, int ldc,
          long sA, long sB, long sC) {
  int bz = blockIdx.z;
  A += (long)bz * sA;  B += (long)bz * sB;  C += (long)bz * sC;
  int wave = threadIdx.x >> 5;
  int rt = blockIdx.x << 4;
  int ct = (blockIdx.y << 6) + (wave << 4);
  if (ct >= N) return;                       // uniform per-wave exit (EXEC stays all-ones)
  const _Float16* Ap = A + (size_t)rt * lda;
  const _Float16* Bp = BT ? (B + (size_t)ct * ldb) : (B + ct);
  v8f acc = {};
  for (int kk = 0; kk < K; kk += 32) {
    v16h a = load_a_frag(Ap, lda, kk);
    v16h b = BT ? load_bt_frag(Bp, ldb, kk) : load_b_rm_frag(Bp, ldb, kk);
    acc = wmma16(a, b, acc);
  }
  int l = lane_id();
  int col  = ct + (l & 15);
  int row0 = rt + ((l >> 4) << 3);           // lanes<16 -> rows rt..rt+7, lanes>=16 -> rt+8..rt+15
  float bv = bias ? bias[col] : 0.f;
  #pragma unroll
  for (int r = 0; r < 8; ++r) {
    float v = acc[r] + bv;
    if (ACT == 1) v = leaky(v);
    C[(size_t)(row0 + r) * ldc + col] = v;
  }
}

// ---------------- conv1 (K=24) + leaky + maxpool4 -> f16 ----------------
__global__ void __launch_bounds__(128)
conv1_pool(const float* __restrict__ x, const float* __restrict__ w1,
           const float* __restrict__ b1, _Float16* __restrict__ A1) {
  int n = blockIdx.x;                        // 0..2047
  __shared__ float xs[WIN_ * SCH_];          // 768 floats
  for (int i = threadIdx.x; i < WIN_ * SCH_; i += 128) xs[i] = x[(size_t)n * WIN_ * SCH_ + i];
  __syncthreads();
  int c = threadIdx.x;
  float w[F_ * SCH_];
  #pragma unroll
  for (int i = 0; i < F_ * SCH_; ++i) w[i] = w1[(size_t)i * CCH_ + c];
  float bc = b1[c];
  for (int t1 = 0; t1 < L1_; ++t1) {
    float mx = -1e30f;
    #pragma unroll
    for (int q = 0; q < 4; ++q) {
      int p0 = 4 * t1 + q;
      float acc = 0.f;
      #pragma unroll
      for (int f = 0; f < F_; ++f)
        #pragma unroll
        for (int s = 0; s < SCH_; ++s)
          acc += xs[(p0 + f) * SCH_ + s] * w[f * SCH_ + s];
      mx = fmaxf(mx, acc);
    }
    A1[((size_t)n * L1_ + t1) * CCH_ + c] = (_Float16)leaky(mx + bc);
  }
}

// ---------------- conv2 implicit-GEMM WMMA + leaky + maxpool4 -> H (2048,13,128) f16 ----------------
// A row (n,t): A1 + (n*62+t)*128, lda=128 (overlapping rows), K=1024 (tap*128+cin)
__global__ void __launch_bounds__(128)
conv2_pool(const _Float16* __restrict__ A1, const _Float16* __restrict__ B2t,
           const float* __restrict__ bias, _Float16* __restrict__ H) {
  int n  = blockIdx.x;
  int t0 = blockIdx.y << 4;                  // 0,16,32,48
  int wave = threadIdx.x >> 5;
  int c0 = wave << 5;                        // each wave: 2 N-tiles (32 channels)
  const _Float16* Ap = A1 + ((size_t)n * L1_ + t0) * CCH_;
  const _Float16* B0 = B2t + (size_t)c0 * 1024;
  const _Float16* B1 = B2t + (size_t)(c0 + 16) * 1024;
  v8f acc0 = {}, acc1 = {};
  for (int kk = 0; kk < 1024; kk += 32) {
    v16h a  = load_a_frag(Ap, CCH_, kk);
    v16h b0 = load_bt_frag(B0, 1024, kk);
    v16h b1 = load_bt_frag(B1, 1024, kk);
    acc0 = wmma16(a, b0, acc0);
    acc1 = wmma16(a, b1, acc1);
  }
  int l = lane_id();
  int hi = l >> 4;
  int t2a  = (t0 >> 2) + hi * 2;             // pooled index for rows 0..3 (this half)
  int col0 = c0 + (l & 15);
  int col1 = col0 + 16;
  float bv0 = bias[col0], bv1 = bias[col1];
  if (t2a < L2_) {
    float p0 = fmaxf(fmaxf(acc0[0], acc0[1]), fmaxf(acc0[2], acc0[3]));
    float p1 = fmaxf(fmaxf(acc1[0], acc1[1]), fmaxf(acc1[2], acc1[3]));
    H[((size_t)n * L2_ + t2a) * CCH_ + col0] = (_Float16)leaky(p0 + bv0);
    H[((size_t)n * L2_ + t2a) * CCH_ + col1] = (_Float16)leaky(p1 + bv1);
  }
  if (t2a + 1 < L2_) {
    float p0 = fmaxf(fmaxf(acc0[4], acc0[5]), fmaxf(acc0[6], acc0[7]));
    float p1 = fmaxf(fmaxf(acc1[4], acc1[5]), fmaxf(acc1[6], acc1[7]));
    H[((size_t)n * L2_ + t2a + 1) * CCH_ + col0] = (_Float16)leaky(p0 + bv0);
    H[((size_t)n * L2_ + t2a + 1) * CCH_ + col1] = (_Float16)leaky(p1 + bv1);
  }
}

// ---------------- GRU sequential scan (keras reset_after=True), fwd+bwd in one launch ----------------
__global__ void __launch_bounds__(512)
gru_scan(const float* __restrict__ xgF, const float* __restrict__ xgB,
         const float* __restrict__ UF,  const float* __restrict__ UB,
         const float* __restrict__ brF, const float* __restrict__ brB,
         const float* __restrict__ h0F, const float* __restrict__ h0B,
         const int* __restrict__ mask, float* __restrict__ Y) {
  bool bw = blockIdx.x != 0;
  const float* xg = bw ? xgB : xgF;
  const float* U  = bw ? UB  : UF;
  const float* br = bw ? brB : brF;
  const float* h0 = bw ? h0B : h0F;
  int coff = bw ? RNN_ : 0;
  __shared__ float hsh[B_ * RNN_];
  int tid = threadIdx.x;
  int b = tid >> 7, j = tid & (RNN_ - 1);
  hsh[tid] = h0[tid];
  __syncthreads();
  for (int step = 0; step < SL_; ++step) {
    int t = bw ? (SL_ - 1 - step) : step;
    float a0 = br[j], a1 = br[j + RNN_], a2 = br[j + 2 * RNN_];
    const float* hb = hsh + b * RNN_;
    for (int k = 0; k < RNN_; ++k) {
      float hv = hb[k];
      const float* Uk = U + (size_t)k * G3_;
      a0 += hv * Uk[j];
      a1 += hv * Uk[j + RNN_];
      a2 += hv * Uk[j + 2 * RNN_];
    }
    const float* xt = xg + (size_t)(b * SL_ + t) * G3_;
    float z  = 1.f / (1.f + __expf(-(xt[j] + a0)));
    float r  = 1.f / (1.f + __expf(-(xt[j + RNN_] + a1)));
    float hh = tanhf(xt[j + 2 * RNN_] + r * a2);
    float hold = hb[j];
    float hn = z * hold + (1.f - z) * hh;
    if (mask[b * SL_ + t] == 0) hn = hold;
    __syncthreads();
    hsh[tid] = hn;
    Y[(size_t)(b * SL_ + t) * (2 * RNN_) + coff + j] = hn;
    __syncthreads();
  }
}

// ---------------- masked softmax over keys -> f16 probs ----------------
__global__ void __launch_bounds__(256)
softmax_mask(const float* __restrict__ S, const int* __restrict__ mask,
             _Float16* __restrict__ P) {
  int row = blockIdx.x;                      // b*512 + t
  int b = row >> 9;
  const float* s = S + (size_t)row * SL_;
  const int* mb = mask + b * SL_;
  __shared__ float red[256];
  __shared__ float vals[SL_];
  int tid = threadIdx.x;
  float mx = -1e30f;
  for (int i = tid; i < SL_; i += 256) {
    float v = mb[i] ? s[i] : -1e9f;
    vals[i] = v;
    mx = fmaxf(mx, v);
  }
  red[tid] = mx; __syncthreads();
  for (int o = 128; o > 0; o >>= 1) { if (tid < o) red[tid] = fmaxf(red[tid], red[tid + o]); __syncthreads(); }
  mx = red[0]; __syncthreads();
  float sum = 0.f;
  for (int i = tid; i < SL_; i += 256) { float e = __expf(vals[i] - mx); vals[i] = e; sum += e; }
  red[tid] = sum; __syncthreads();
  for (int o = 128; o > 0; o >>= 1) { if (tid < o) red[tid] += red[tid + o]; __syncthreads(); }
  float inv = 1.f / red[0];
  for (int i = tid; i < SL_; i += 256) P[(size_t)row * SL_ + i] = (_Float16)(vals[i] * inv);
}

// ---------------- small utility kernels ----------------
__global__ void f32_to_f16(const float* __restrict__ in, _Float16* __restrict__ o, int n) {
  int i = blockIdx.x * 256 + threadIdx.x;
  if (i < n) o[i] = (_Float16)in[i];
}
// in (K,N) f32 -> out (N,K) f16
__global__ void transpose_to_f16(const float* __restrict__ in, _Float16* __restrict__ o, int K, int N) {
  int i = blockIdx.x * 256 + threadIdx.x;
  if (i >= K * N) return;
  int k = i / N, c = i - k * N;
  o[(size_t)c * K + k] = (_Float16)in[i];
}
// per-row mask (query mask zeroing) + convert
__global__ void mask_to_f16(const float* __restrict__ in, const int* __restrict__ m,
                            _Float16* __restrict__ o, int rows, int cols) {
  int i = blockIdx.x * 256 + threadIdx.x;
  if (i >= rows * cols) return;
  int r = i / cols;
  o[i] = (_Float16)(m[r] ? in[i] : 0.f);
}
// final dense: out[r] = X[r,:]·w + b
__global__ void __launch_bounds__(256)
dense3(const float* __restrict__ X, const float* __restrict__ w,
       const float* __restrict__ b, float* __restrict__ out) {
  int r = blockIdx.x, tid = threadIdx.x;
  __shared__ float red[256];
  red[tid] = X[(size_t)r * FC_ + tid] * w[tid];
  __syncthreads();
  for (int o = 128; o > 0; o >>= 1) { if (tid < o) red[tid] += red[tid + o]; __syncthreads(); }
  if (tid == 0) out[r] = red[0] + b[0];
}

// ---------------- host orchestration ----------------
extern "C" void kernel_launch(void* const* d_in, const int* in_sizes, int n_in,
                              void* d_out, int out_size, void* d_ws, size_t ws_size,
                              hipStream_t stream) {
  (void)in_sizes; (void)n_in; (void)out_size;
  const float* x    = (const float*)d_in[0];
  const int*   m    = (const int*)  d_in[1];
  const float* s1f  = (const float*)d_in[2];
  const float* s1b  = (const float*)d_in[3];
  const float* s2f  = (const float*)d_in[4];
  const float* s2b  = (const float*)d_in[5];
  const float* w1   = (const float*)d_in[6];
  const float* b1   = (const float*)d_in[7];
  const float* w2   = (const float*)d_in[8];
  const float* b2   = (const float*)d_in[9];
  const float* g1fW = (const float*)d_in[10]; const float* g1fU = (const float*)d_in[11]; const float* g1fb = (const float*)d_in[12];
  const float* g1bW = (const float*)d_in[13]; const float* g1bU = (const float*)d_in[14]; const float* g1bb = (const float*)d_in[15];
  const float* g2fW = (const float*)d_in[16]; const float* g2fU = (const float*)d_in[17]; const float* g2fb = (const float*)d_in[18];
  const float* g2bW = (const float*)d_in[19]; const float* g2bU = (const float*)d_in[20]; const float* g2bb = (const float*)d_in[21];
  const float* d1w  = (const float*)d_in[22]; const float* d1b  = (const float*)d_in[23];
  const float* d2w  = (const float*)d_in[24]; const float* d2b  = (const float*)d_in[25];
  const float* d3w  = (const float*)d_in[26]; const float* d3b  = (const float*)d_in[27];
  float* out = (float*)d_out;

  char* ws = (char*)d_ws; size_t off = 0;
  auto alloc = [&](size_t bytes) -> void* {
    off = (off + 255) & ~(size_t)255;
    void* p = ws + off; off += bytes; return p;
  };
  _Float16* A1   = (_Float16*)alloc(((size_t)NSEQ * L1_ + 16) * CCH_ * 2);  // +16 pad rows
  _Float16* B2t  = (_Float16*)alloc((size_t)CCH_ * 1024 * 2);
  _Float16* Hf   = (_Float16*)alloc((size_t)NSEQ * D1_ * 2);
  _Float16* W1ft = (_Float16*)alloc((size_t)G3_ * D1_ * 2);
  _Float16* W1bt = (_Float16*)alloc((size_t)G3_ * D1_ * 2);
  _Float16* W2ft = (_Float16*)alloc((size_t)G3_ * (2*RNN_) * 2);
  _Float16* W2bt = (_Float16*)alloc((size_t)G3_ * (2*RNN_) * 2);
  _Float16* D1t  = (_Float16*)alloc((size_t)FC_ * (2*RNN_) * 2);
  _Float16* D2t  = (_Float16*)alloc((size_t)FC_ * FC_ * 2);
  float* xg1f = (float*)alloc((size_t)NSEQ * G3_ * 4);
  float* xg1b = (float*)alloc((size_t)NSEQ * G3_ * 4);
  float* xg2f = (float*)alloc((size_t)NSEQ * G3_ * 4);
  float* xg2b = (float*)alloc((size_t)NSEQ * G3_ * 4);
  float*    h1   = (float*)   alloc((size_t)NSEQ * 2*RNN_ * 4);
  _Float16* h1f  = (_Float16*)alloc((size_t)NSEQ * 2*RNN_ * 2);
  float*    h2   = (float*)   alloc((size_t)NSEQ * 2*RNN_ * 4);
  _Float16* h2f  = (_Float16*)alloc((size_t)NSEQ * 2*RNN_ * 2);
  float*    sc   = (float*)   alloc((size_t)B_ * SL_ * SL_ * 4);
  _Float16* P    = (_Float16*)alloc((size_t)B_ * SL_ * SL_ * 2);
  float*    attn = (float*)   alloc((size_t)NSEQ * 2*RNN_ * 4);
  _Float16* attnf= (_Float16*)alloc((size_t)NSEQ * 2*RNN_ * 2);
  float*    d1o  = (float*)   alloc((size_t)NSEQ * FC_ * 4);
  _Float16* d1f  = (_Float16*)alloc((size_t)NSEQ * FC_ * 2);
  float*    d2o  = (float*)   alloc((size_t)NSEQ * FC_ * 4);
  if (off > ws_size) return;   // workspace too small: bail deterministically

  auto T = [&](const float* in, _Float16* o, int K, int N) {
    int tot = K * N;
    transpose_to_f16<<<(tot + 255) / 256, 256, 0, stream>>>(in, o, K, N);
  };
  // weight prep (f16, transposed to (N,K))
  T(w2,   B2t,  1024, CCH_);
  T(g1fW, W1ft, D1_,  G3_);
  T(g1bW, W1bt, D1_,  G3_);
  T(g2fW, W2ft, 2*RNN_, G3_);
  T(g2bW, W2bt, 2*RNN_, G3_);
  T(d1w,  D1t,  2*RNN_, FC_);
  T(d2w,  D2t,  FC_,  FC_);

  // conv1 + pool -> A1 (f16)
  conv1_pool<<<NSEQ, 128, 0, stream>>>(x, w1, b1, A1);
  // conv2 (WMMA implicit GEMM) + pool -> H (2048,1664) f16
  conv2_pool<<<dim3(NSEQ, 4), 128, 0, stream>>>(A1, B2t, b2, Hf);

  // GRU1 input projections: (2048,1664)@(1664,384)+b
  wmma_gemm<true,0><<<dim3(NSEQ/16, G3_/64, 1), 128, 0, stream>>>(
      Hf, W1ft, g1fb, xg1f, G3_, D1_, D1_, D1_, G3_, 0, 0, 0);
  wmma_gemm<true,0><<<dim3(NSEQ/16, G3_/64, 1), 128, 0, stream>>>(
      Hf, W1bt, g1bb, xg1b, G3_, D1_, D1_, D1_, G3_, 0, 0, 0);
  // GRU1 scan (fwd + bwd)
  gru_scan<<<2, 512, 0, stream>>>(xg1f, xg1b, g1fU, g1bU,
                                  g1fb + G3_, g1bb + G3_, s1f, s1b, m, h1);
  f32_to_f16<<<(NSEQ*2*RNN_ + 255)/256, 256, 0, stream>>>(h1, h1f, NSEQ*2*RNN_);

  // GRU2 input projections: (2048,256)@(256,384)+b
  wmma_gemm<true,0><<<dim3(NSEQ/16, G3_/64, 1), 128, 0, stream>>>(
      h1f, W2ft, g2fb, xg2f, G3_, 2*RNN_, 2*RNN_, 2*RNN_, G3_, 0, 0, 0);
  wmma_gemm<true,0><<<dim3(NSEQ/16, G3_/64, 1), 128, 0, stream>>>(
      h1f, W2bt, g2bb, xg2b, G3_, 2*RNN_, 2*RNN_, 2*RNN_, G3_, 0, 0, 0);
  gru_scan<<<2, 512, 0, stream>>>(xg2f, xg2b, g2fU, g2bU,
                                  g2fb + G3_, g2bb + G3_, s2f, s2b, m, h2);
  f32_to_f16<<<(NSEQ*2*RNN_ + 255)/256, 256, 0, stream>>>(h2, h2f, NSEQ*2*RNN_);

  // attention scores: per batch (512,256)@(512,256)^T  (h2f is both A and Bt)
  wmma_gemm<true,0><<<dim3(SL_/16, SL_/64, B_), 128, 0, stream>>>(
      h2f, h2f, nullptr, sc, SL_, 2*RNN_, 2*RNN_, 2*RNN_, SL_,
      (long)SL_*2*RNN_, (long)SL_*2*RNN_, (long)SL_*SL_);
  softmax_mask<<<NSEQ, 256, 0, stream>>>(sc, m, P);
  // attention output: P(512,512) @ h2(512,256), B row-major (K,N)
  wmma_gemm<false,0><<<dim3(SL_/16, (2*RNN_)/64, B_), 128, 0, stream>>>(
      P, h2f, nullptr, attn, 2*RNN_, SL_, SL_, 2*RNN_, 2*RNN_,
      (long)SL_*SL_, (long)SL_*2*RNN_, (long)SL_*2*RNN_);
  // query-mask zeroing + f16
  mask_to_f16<<<(NSEQ*2*RNN_ + 255)/256, 256, 0, stream>>>(attn, m, attnf, NSEQ, 2*RNN_);

  // dense1 (leaky) -> dense2 (leaky) -> dense3
  wmma_gemm<true,1><<<dim3(NSEQ/16, FC_/64, 1), 128, 0, stream>>>(
      attnf, D1t, d1b, d1o, FC_, 2*RNN_, 2*RNN_, 2*RNN_, FC_, 0, 0, 0);
  f32_to_f16<<<(NSEQ*FC_ + 255)/256, 256, 0, stream>>>(d1o, d1f, NSEQ*FC_);
  wmma_gemm<true,1><<<dim3(NSEQ/16, FC_/64, 1), 128, 0, stream>>>(
      d1f, D2t, d2b, d2o, FC_, FC_, FC_, FC_, FC_, 0, 0, 0);
  dense3<<<NSEQ, 256, 0, stream>>>(d2o, d3w, d3b, out);
}
